// MinimalRNNAttention_33595234189740
// MI455X (gfx1250) — matrix operand order repaired and checked by reading, not compiled, not run on hardware
//
#include <hip/hip_runtime.h>
#include <math.h>

#define B_   4
#define S_   2048
#define E_   1024
#define H_   8
#define D_   128
#define T_   (B_*S_)    // 8192 tokens
#define HD_  (H_*D_)    // 1024
#define EPSV 1e-5f

typedef __attribute__((ext_vector_type(16))) __bf16 v16bf;
typedef __attribute__((ext_vector_type(8)))  float  v8f;

union FragAB { v16bf v; unsigned short us[16]; uint4 q[2]; };

__device__ __forceinline__ unsigned short f2bf(float f) {
    unsigned u = __float_as_uint(f);
    unsigned r = u + 0x7FFFu + ((u >> 16) & 1u);   // round-to-nearest-even
    return (unsigned short)(r >> 16);
}

__device__ __forceinline__ v8f wmma_bf16(v16bf a, v16bf b, v8f c) {
    return __builtin_amdgcn_wmma_f32_16x16x32_bf16(false, a, false, b, (short)0, c, false, false);
}

// A fragment: 16x32 tile of a row-major bf16 matrix (lane&15 = row, lane>>4 selects K half)
// Per lane: two contiguous 8-element runs at K = k0+8g and K = k0+16+8g (ISA 7.12.2 16-bit A layout)
__device__ __forceinline__ v16bf load_a_bf16(const unsigned short* base, int ldk, int m0, int k0, int lane) {
    int r = m0 + (lane & 15);
    int g = lane >> 4;
    const unsigned short* p = base + (size_t)r * ldk + k0 + 8 * g;
    FragAB f;
    f.q[0] = *(const uint4*)p;          // K = k0+8g .. +7
    f.q[1] = *(const uint4*)(p + 16);   // K = k0+16+8g .. +7
    return f.v;
}

// B fragment: 32x16 tile from K-contiguous (pre-transposed [N][K]) bf16 weights.
// Per lane (col n0+(lane&15)): two 8-element runs at K = k0+8g and K = k0+16+8g (matches A).
__device__ __forceinline__ v16bf load_b_bf16(const unsigned short* baseT, int ldk, int n0, int k0, int lane) {
    int n = n0 + (lane & 15);
    int g = lane >> 4;
    const unsigned short* p = baseT + (size_t)n * ldk + k0 + 8 * g;
    FragAB f;
    f.q[0] = *(const uint4*)p;
    f.q[1] = *(const uint4*)(p + 16);
    return f.v;
}

// ---------------- prep kernels: activation/weight convert + transpose ----------------

// x f32 -> bf16 row-major (keeps the GEMM inner loop free of cvt VALU work)
__global__ void k_prep_xbf(const float* __restrict__ x, unsigned short* __restrict__ xbf) {
    int idx = blockIdx.x * 256 + threadIdx.x;          // T*E/4 = 2M threads, 4 elems each
    float4 v = *(const float4*)(x + (size_t)idx * 4);
    ushort4 o;
    o.x = f2bf(v.x); o.y = f2bf(v.y); o.z = f2bf(v.z); o.w = f2bf(v.w);
    *(ushort4*)(xbf + (size_t)idx * 4) = o;
}

__global__ void k_prep_wezt(const float* __restrict__ W, unsigned short* __restrict__ Wt) {
    int idx = blockIdx.x * 256 + threadIdx.x;          // E*HD = 1M, src coalesced
    int k = idx >> 10, n = idx & 1023;                 // W[k][n], k=E-dim
    Wt[(size_t)n * 1024 + k] = f2bf(W[idx]);
}

__global__ void k_prep_w1t(const float* __restrict__ W1, unsigned short* __restrict__ W1t) {
    int idx = blockIdx.x * 256 + threadIdx.x;          // H*D*D = 131072
    int h = idx >> 14;
    int d = (idx >> 7) & 127;
    int e = idx & 127;                                 // src = ff_W1[h][d][e]
    W1t[(size_t)h * 16384 + (size_t)e * 128 + d] = f2bf(W1[idx]);
}

__global__ void k_prep_w2t(const float* __restrict__ W2, unsigned short* __restrict__ W2t) {
    int idx = blockIdx.x * 256 + threadIdx.x;          // H*D*E = 1M
    int h = idx >> 17;
    int d = (idx >> 10) & 127;
    int e = idx & 1023;                                // src = ff_W2[h][d][e]
    W2t[(size_t)e * 1024 + h * 128 + d] = f2bf(W2[idx]);
}

// v[h*128+d] = ff_W2[h][d][:] . w_att ; c[h] = ff_b2[h][:] . w_att + b_att
__global__ void k_prep_vc(const float* __restrict__ W2, const float* __restrict__ b2,
                          const float* __restrict__ w_att, const float* __restrict__ b_att,
                          float* __restrict__ vvec, float* __restrict__ cvec) {
    int gid = blockIdx.x * 256 + threadIdx.x;          // 1032 rows * 32 lanes = 129 blocks
    int row = gid >> 5, lane = gid & 31;
    float acc = 0.f;
    if (row < 1024) {
        int h = row >> 7, d = row & 127;
        const float* src = W2 + (size_t)h * 131072 + (size_t)d * 1024;
        for (int e = lane; e < 1024; e += 32) acc += src[e] * w_att[e];
    } else {
        int h = row - 1024;
        const float* src = b2 + (size_t)h * 1024;
        for (int e = lane; e < 1024; e += 32) acc += src[e] * w_att[e];
    }
    for (int off = 16; off; off >>= 1) acc += __shfl_xor(acc, off, 32);
    if (lane == 0) {
        if (row < 1024) vvec[row] = acc;
        else            cvec[row - 1024] = acc + b_att[0];
    }
}

// ---------------- kernel 1: z = tanh(x @ W_ez + b_ez), M=8192 K=1024 N=1024 ----------------
// Workgroup = 8 waves: 2 M-tiles x 4 wave-cols of 128 cols each -> 32x512 output tile.
// A fragment amortized over 8 WMMAs per wave per K-step.
__global__ __launch_bounds__(256) void k_gemm_z(const unsigned short* __restrict__ xbf,
        const unsigned short* __restrict__ Wt, const float* __restrict__ b_ez,
        float* __restrict__ z) {
    int lane = threadIdx.x & 31;
    int w    = threadIdx.x >> 5;
    int m0   = blockIdx.x * 32 + (w & 1) * 16;
    int n0w  = blockIdx.y * 512 + (w >> 1) * 128;
    v8f acc[8];
    for (int t = 0; t < 8; ++t) for (int i = 0; i < 8; ++i) acc[t][i] = 0.f;
    for (int kk = 0; kk < 1024; kk += 32) {
        if (kk + 32 < 1024)
            __builtin_prefetch(xbf + (size_t)(m0 + (lane & 15)) * 1024 + kk + 32, 0, 1);
        v16bf a = load_a_bf16(xbf, 1024, m0, kk, lane);
#pragma unroll
        for (int t = 0; t < 8; ++t) {
            v16bf b = load_b_bf16(Wt, 1024, n0w + 16 * t, kk, lane);
            acc[t] = wmma_bf16(a, b, acc[t]);
        }
    }
    int g = lane >> 4;
#pragma unroll
    for (int t = 0; t < 8; ++t) {
        int ncol = n0w + 16 * t + (lane & 15);
        float bias = b_ez[ncol];
#pragma unroll
        for (int i = 0; i < 8; ++i) {
            int row = m0 + i + 8 * g;
            z[(size_t)row * 1024 + ncol] = tanhf(acc[t][i] + bias);
        }
    }
}

// ---------------- kernel 2: recurrent scan, one wave per (b,h) sequence ----------------
// Fuses: gated update, state LayerNorm, os_diag scale, per-head ff LayerNorm -> bf16 hln
__global__ __launch_bounds__(256) void k_scan(const float* __restrict__ z,
        const float* __restrict__ U_h, const float* __restrict__ U_z,
        const float* __restrict__ b_u, const float* __restrict__ lns_g, const float* __restrict__ lns_b,
        const float* __restrict__ out_shaper, const float* __restrict__ ff_ln_g,
        const float* __restrict__ ff_ln_b, unsigned short* __restrict__ hln) {
    int lane = threadIdx.x & 31;
    int seq  = (blockIdx.x * 256 + threadIdx.x) >> 5;
    if (seq >= B_ * H_) return;
    int b = seq >> 3, h = seq & 7;
    int d0 = lane * 4;
    float uh[4], uz[4], osd[4], bu[4], lg[4], lb[4], fg[4], fb[4];
#pragma unroll
    for (int j = 0; j < 4; ++j) {
        int d = d0 + j;
        uh[j]  = U_h[(size_t)h * 16384 + (size_t)d * 128 + d];          // diagonal
        uz[j]  = U_z[(size_t)h * 16384 + (size_t)d * 128 + d];
        osd[j] = out_shaper[(size_t)h * 16384 + (size_t)d * 128 + d];
        bu[j]  = b_u[h * 128 + d];
        lg[j]  = lns_g[d];              lb[j] = lns_b[d];
        fg[j]  = ff_ln_g[h * 128 + d];  fb[j] = ff_ln_b[h * 128 + d];
    }
    float hp[4] = {0.f, 0.f, 0.f, 0.f};
    const float inv = 1.0f / 128.0f;
    for (int t = 0; t < S_; ++t) {
        size_t rowoff = ((size_t)(b * S_ + t)) * 1024 + h * 128 + d0;
        float4 zt4 = *(const float4*)(z + rowoff);
        float zt[4] = {zt4.x, zt4.y, zt4.z, zt4.w};
        float hn[4], s = 0.f, q = 0.f;
#pragma unroll
        for (int j = 0; j < 4; ++j) {
            float a = hp[j] * uh[j] + zt[j] * uz[j] + bu[j];
            float u = 1.0f / (1.0f + __expf(-a));
            hn[j] = u * hp[j] + (1.0f - u) * zt[j];
            s += hn[j]; q += hn[j] * hn[j];
        }
#pragma unroll
        for (int off = 16; off; off >>= 1) { s += __shfl_xor(s, off, 32); q += __shfl_xor(q, off, 32); }
        float mean = s * inv, var = q * inv - mean * mean;
        float rs = rsqrtf(var + EPSV);
        float sp[4]; float s2 = 0.f, q2 = 0.f;
#pragma unroll
        for (int j = 0; j < 4; ++j) {
            hp[j] = (hn[j] - mean) * rs * lg[j] + lb[j];   // carried state (post-LN)
            sp[j] = hp[j] * osd[j];                        // os_diag shaping
            s2 += sp[j]; q2 += sp[j] * sp[j];
        }
#pragma unroll
        for (int off = 16; off; off >>= 1) { s2 += __shfl_xor(s2, off, 32); q2 += __shfl_xor(q2, off, 32); }
        float m2 = s2 * inv, v2 = q2 * inv - m2 * m2;
        float r2 = rsqrtf(v2 + EPSV);
        unsigned px = 0, py = 0;
#pragma unroll
        for (int j = 0; j < 4; ++j) {
            float o = (sp[j] - m2) * r2 * fg[j] + fb[j];
            unsigned bb = f2bf(o);
            if (j == 0) px = bb; else if (j == 1) px |= bb << 16;
            else if (j == 2) py = bb; else py |= bb << 16;
        }
        uint2 pk; pk.x = px; pk.y = py;
        *(uint2*)(hln + rowoff) = pk;
    }
}

// ---------------- kernel 3: per-head FFN stage 1 + GELU + attention softmax -> g ----------------
// Workgroup = 16 tokens; wave w handles head w: h1 = gelu(hln_h @ W1_h + b1_h),
// score_h = h1.v_h + c_h; softmax over heads; g = s_h * h1 (bf16).
__global__ __launch_bounds__(256) void k_ffn1(const unsigned short* __restrict__ hln,
        const unsigned short* __restrict__ W1t, const float* __restrict__ ff_b1,
        const float* __restrict__ vvec, const float* __restrict__ cvec,
        unsigned short* __restrict__ gbuf, float* __restrict__ sws) {
    __shared__ float sc[16][8];
    __shared__ float sm[16][8];
    int lane = threadIdx.x & 31;
    int w    = threadIdx.x >> 5;          // head index
    int m0   = blockIdx.x * 16;
    int g    = lane >> 4;
    v8f acc[8];
    for (int t = 0; t < 8; ++t) for (int i = 0; i < 8; ++i) acc[t][i] = 0.f;
    const unsigned short* abase = hln + w * 128;
    const unsigned short* bbase = W1t + (size_t)w * 16384;
    for (int kk = 0; kk < 128; kk += 32) {
        v16bf a = load_a_bf16(abase, 1024, m0, kk, lane);
#pragma unroll
        for (int t = 0; t < 8; ++t) {
            v16bf b = load_b_bf16(bbase, 128, t * 16, kk, lane);
            acc[t] = wmma_bf16(a, b, acc[t]);
        }
    }
    float rsum[8];
#pragma unroll
    for (int i = 0; i < 8; ++i) rsum[i] = 0.f;
#pragma unroll
    for (int t = 0; t < 8; ++t) {
        int ncol = t * 16 + (lane & 15);
        float bias = ff_b1[w * 128 + ncol];
        float vv   = vvec[w * 128 + ncol];
#pragma unroll
        for (int i = 0; i < 8; ++i) {
            float xg = acc[t][i] + bias;
            float ge = 0.5f * xg * (1.0f + erff(xg * 0.70710678118f));  // exact GELU
            acc[t][i] = ge;
            rsum[i] += ge * vv;
        }
    }
#pragma unroll
    for (int i = 0; i < 8; ++i) {       // reduce over 16-lane column group -> per-token dot
        float r = rsum[i];
        r += __shfl_xor(r, 1, 32); r += __shfl_xor(r, 2, 32);
        r += __shfl_xor(r, 4, 32); r += __shfl_xor(r, 8, 32);
        if ((lane & 15) == 0) sc[i + 8 * g][w] = r + cvec[w];
    }
    __syncthreads();
    if (threadIdx.x < 16) {             // softmax over 8 heads for one token each
        int r = threadIdx.x;
        float mx = -1e30f;
        for (int hh = 0; hh < 8; ++hh) mx = fmaxf(mx, sc[r][hh]);
        float e[8], ssum = 0.f;
        for (int hh = 0; hh < 8; ++hh) { e[hh] = __expf(sc[r][hh] - mx); ssum += e[hh]; }
        float rinv = 1.0f / ssum;
        for (int hh = 0; hh < 8; ++hh) {
            float sv = e[hh] * rinv;
            sm[r][hh] = sv;
            sws[(size_t)(m0 + r) * 8 + hh] = sv;
        }
    }
    __syncthreads();
#pragma unroll
    for (int t = 0; t < 8; ++t) {
        int ncol = t * 16 + (lane & 15);
#pragma unroll
        for (int i = 0; i < 8; ++i) {
            int r = i + 8 * g;
            float val = acc[t][i] * sm[r][w];
            gbuf[(size_t)(m0 + r) * 1024 + w * 128 + ncol] = f2bf(val);
        }
    }
}

// ---------------- kernel 4: weighted = g @ W2cat + sum_h s_h*b2_h, then output LayerNorm ----------------
// Workgroup = 16 tokens x full E=1024 row (wave w covers cols w*128..+127) so LN fuses in.
__global__ __launch_bounds__(256) void k_gemm2_ln(const unsigned short* __restrict__ gbuf,
        const unsigned short* __restrict__ W2t, const float* __restrict__ ff_b2,
        const float* __restrict__ sws, const float* __restrict__ lno_g, const float* __restrict__ lno_b,
        float* __restrict__ out) {
    __shared__ float sl[16][8];
    __shared__ float red[16][2];
    int lane = threadIdx.x & 31;
    int w    = threadIdx.x >> 5;
    int m0   = blockIdx.x * 16;
    int g    = lane >> 4;
    if (threadIdx.x < 128) {
        int r = threadIdx.x >> 3, hh = threadIdx.x & 7;
        sl[r][hh] = sws[(size_t)(m0 + r) * 8 + hh];
    }
    if (threadIdx.x < 32) red[threadIdx.x >> 1][threadIdx.x & 1] = 0.f;
    __syncthreads();
    v8f acc[8];
    for (int t = 0; t < 8; ++t) for (int i = 0; i < 8; ++i) acc[t][i] = 0.f;
    for (int kk = 0; kk < 1024; kk += 32) {
        if (kk + 32 < 1024)
            __builtin_prefetch(gbuf + (size_t)(m0 + (lane & 15)) * 1024 + kk + 32, 0, 1);
        v16bf a = load_a_bf16(gbuf, 1024, m0, kk, lane);
#pragma unroll
        for (int t = 0; t < 8; ++t) {
            v16bf b = load_b_bf16(W2t, 1024, w * 128 + t * 16, kk, lane);
            acc[t] = wmma_bf16(a, b, acc[t]);
        }
    }
    float s1[8], s2[8];
#pragma unroll
    for (int i = 0; i < 8; ++i) { s1[i] = 0.f; s2[i] = 0.f; }
#pragma unroll
    for (int t = 0; t < 8; ++t) {
        int nglob = w * 128 + t * 16 + (lane & 15);
        float b2v[8];
#pragma unroll
        for (int hh = 0; hh < 8; ++hh) b2v[hh] = ff_b2[hh * 1024 + nglob];
#pragma unroll
        for (int i = 0; i < 8; ++i) {
            int r = i + 8 * g;
            float val = acc[t][i];
#pragma unroll
            for (int hh = 0; hh < 8; ++hh) val += sl[r][hh] * b2v[hh];   // softmax-weighted bias
            acc[t][i] = val;
            s1[i] += val; s2[i] += val * val;
        }
    }
#pragma unroll
    for (int i = 0; i < 8; ++i) {       // column-group reduce then cross-wave LDS accumulate
        float a1 = s1[i], a2 = s2[i];
        a1 += __shfl_xor(a1, 1, 32); a2 += __shfl_xor(a2, 1, 32);
        a1 += __shfl_xor(a1, 2, 32); a2 += __shfl_xor(a2, 2, 32);
        a1 += __shfl_xor(a1, 4, 32); a2 += __shfl_xor(a2, 4, 32);
        a1 += __shfl_xor(a1, 8, 32); a2 += __shfl_xor(a2, 8, 32);
        if ((lane & 15) == 0) {
            atomicAdd(&red[i + 8 * g][0], a1);
            atomicAdd(&red[i + 8 * g][1], a2);
        }
    }
    __syncthreads();
    float mean[8], rstd[8];
    const float inv = 1.0f / 1024.0f;
#pragma unroll
    for (int i = 0; i < 8; ++i) {
        int r = i + 8 * g;
        float m = red[r][0] * inv;
        float vv = red[r][1] * inv - m * m;
        mean[i] = m; rstd[i] = rsqrtf(vv + EPSV);
    }
#pragma unroll
    for (int t = 0; t < 8; ++t) {
        int nglob = w * 128 + t * 16 + (lane & 15);
        float gsc = lno_g[nglob], bsc = lno_b[nglob];
#pragma unroll
        for (int i = 0; i < 8; ++i) {
            int r = i + 8 * g;
            out[(size_t)(m0 + r) * 1024 + nglob] = (acc[t][i] - mean[i]) * rstd[i] * gsc + bsc;
        }
    }
}

extern "C" void kernel_launch(void* const* d_in, const int* in_sizes, int n_in,
                              void* d_out, int out_size, void* d_ws, size_t ws_size,
                              hipStream_t stream) {
    (void)in_sizes; (void)n_in; (void)out_size; (void)ws_size;
    const float* x          = (const float*)d_in[0];
    const float* W_ez       = (const float*)d_in[1];
    const float* b_ez       = (const float*)d_in[2];
    const float* U_h        = (const float*)d_in[3];
    const float* U_z        = (const float*)d_in[4];
    const float* b_u        = (const float*)d_in[5];
    const float* out_shaper = (const float*)d_in[6];
    const float* lns_g      = (const float*)d_in[7];
    const float* lns_b      = (const float*)d_in[8];
    const float* ff_ln_g    = (const float*)d_in[9];
    const float* ff_ln_b    = (const float*)d_in[10];
    const float* ff_W1      = (const float*)d_in[11];
    const float* ff_b1      = (const float*)d_in[12];
    const float* ff_W2      = (const float*)d_in[13];
    const float* ff_b2      = (const float*)d_in[14];
    const float* w_att      = (const float*)d_in[15];
    const float* b_att      = (const float*)d_in[16];
    const float* lno_g      = (const float*)d_in[17];
    const float* lno_b      = (const float*)d_in[18];

    char* ws = (char*)d_ws;
    unsigned short* Wezt = (unsigned short*)(ws + 0);          // 2 MB   [n][k] bf16
    unsigned short* W1t  = (unsigned short*)(ws + 2097152);    // 256 KB [h][e][d] bf16
    unsigned short* W2t  = (unsigned short*)(ws + 2359296);    // 2 MB   [e][h*128+d] bf16
    float*          vvec = (float*)(ws + 4456448);             // 4 KB
    float*          cvec = (float*)(ws + 4460544);             // 32 B (padded)
    float*          zbuf = (float*)(ws + 4464640);             // 32 MB  [T][HD] f32
    unsigned short* gbuf = (unsigned short*)(ws + 4464640);    // 16 MB, aliases zbuf (dead after scan)
    unsigned short* hln  = (unsigned short*)(ws + 38019072);   // 16 MB  [T][HD] bf16
    unsigned short* xbf  = (unsigned short*)(ws + 38019072);   // 16 MB, aliases hln (dead after gemm_z)
    float*          sws  = (float*)(ws + 54796288);            // 256 KB [T][H] f32

    k_prep_xbf <<<8192, 256, 0, stream>>>(x, xbf);             // T*E/4 threads
    k_prep_wezt<<<4096, 256, 0, stream>>>(W_ez, Wezt);
    k_prep_w1t <<<512,  256, 0, stream>>>(ff_W1, W1t);
    k_prep_w2t <<<4096, 256, 0, stream>>>(ff_W2, W2t);
    k_prep_vc  <<<129,  256, 0, stream>>>(ff_W2, ff_b2, w_att, b_att, vvec, cvec);

    k_gemm_z   <<<dim3(T_/32, HD_/512), 256, 0, stream>>>(xbf, Wezt, b_ez, zbuf);
    k_scan     <<<4,   256, 0, stream>>>(zbuf, U_h, U_z, b_u, lns_g, lns_b,
                                         out_shaper, ff_ln_g, ff_ln_b, hln);
    k_ffn1     <<<T_/16, 256, 0, stream>>>(hln, W1t, ff_b1, vvec, cvec, gbuf, sws);
    k_gemm2_ln <<<T_/16, 256, 0, stream>>>(gbuf, W2t, ff_b2, sws, lno_g, lno_b, (float*)d_out);
}
